// GATConv_2946347565058
// MI455X (gfx1250) — matrix operand order repaired and checked by reading, not compiled, not run on hardware
//
#include <hip/hip_runtime.h>
#include <hip/hip_bf16.h>

typedef __attribute__((ext_vector_type(16))) _Float16 v16h;
typedef __attribute__((ext_vector_type(8)))  _Float16 v8h;
typedef __attribute__((ext_vector_type(8)))  float    v8f;

#define N_NODES 50000
#define N_EDGES 800000
#define IN_DIM  128
#define HEADS   4
#define OUT_DIM 32
#define HD      (HEADS * OUT_DIM)   // 128
#define XS_PITCH 136                // halfs; 272 B rows -> 16 B aligned, bank spread

// ---------------------------------------------------------------------------
// Kernel 0: zero the accumulators (d_out and att_sum are poisoned by harness)
// ---------------------------------------------------------------------------
__global__ void zero_kernel(float* __restrict__ out, float* __restrict__ att_sum) {
    int i = blockIdx.x * blockDim.x + threadIdx.x;
    if (i < N_NODES * HD)    out[i]     = 0.0f;
    if (i < N_NODES * HEADS) att_sum[i] = 0.0f;
}

// ---------------------------------------------------------------------------
// Kernel 0b: pack W (f32, 128x128) -> f16 once per launch. All 3125 GEMM
// blocks then read conversion-free, 32 B-aligned B fragments.
// ---------------------------------------------------------------------------
__global__ void pack_w_kernel(const float* __restrict__ W, _Float16* __restrict__ Wh) {
    int i = blockIdx.x * blockDim.x + threadIdx.x;
    if (i < HD * IN_DIM) Wh[i] = (_Float16)W[i];
}

// ---------------------------------------------------------------------------
// Kernel 1: h = x @ W^T via v_wmma_f32_16x16x32_f16 (K fully unrolled: 4 WMMA)
//   block = 256 = 8 waves; block stages one 16x128 x-tile in LDS (as f16),
//   wave w computes output columns [w*16, w*16+16).
//   A frag (16-bit A 16x32): lane l: m=l&15, hl=l>>4; e -> K=hl*16+e  (16 contiguous halfs)
//   B frag (32x16): lane l: n=l&15, hl=l>>4; e -> K=hl*16+e  (16 contiguous halfs)
//   C/D: vgpr v, lane l -> row = 8*(l>>4)+v, col = l&15
// ---------------------------------------------------------------------------
__global__ __launch_bounds__(256) void gemm_h_kernel(const float* __restrict__ x,
                                                     const _Float16* __restrict__ Wh,
                                                     float* __restrict__ h) {
    __shared__ _Float16 xs[16 * XS_PITCH];

    const int rt  = blockIdx.x;        // row tile: rows [rt*16, rt*16+16)
    const int tid = threadIdx.x;

    // stage + convert x tile once: 16 x 128 elements, 8 per thread
    for (int i = tid; i < 16 * IN_DIM; i += 256) {
        int r = i >> 7, c = i & 127;
        xs[r * XS_PITCH + c] = (_Float16)x[(rt * 16 + r) * IN_DIM + c];
    }
    __syncthreads();

    const int wave = tid >> 5;         // column tile 0..7
    const int lane = tid & 31;
    const int hl   = lane >> 4;        // half-wave select
    const int mn   = lane & 15;        // A row / B col within tile

    const _Float16* arow = &xs[mn * XS_PITCH];
    const _Float16* brow = &Wh[(wave * 16 + mn) * IN_DIM];

    v8f c = {};
#pragma unroll
    for (int kk = 0; kk < IN_DIM; kk += 32) {
        v8h alo = *(const v8h*)(arow + kk + hl * 16);       // ds_load_b128
        v8h ahi = *(const v8h*)(arow + kk + hl * 16 + 8);   // ds_load_b128
        v16h a  = __builtin_shufflevector(alo, ahi,
                    0,1,2,3,4,5,6,7,8,9,10,11,12,13,14,15);
        v16h b  = *(const v16h*)(brow + kk + hl * 16);      // 32 B global load
        c = __builtin_amdgcn_wmma_f32_16x16x32_f16(false, a, false, b,
                                                   (short)0, c, false, false);
    }

#pragma unroll
    for (int v = 0; v < 8; ++v) {
        int row = rt * 16 + hl * 8 + v;
        int col = wave * 16 + mn;
        h[row * HD + col] = c[v];
    }
}

// ---------------------------------------------------------------------------
// Kernel 2: per-node attention logits
//   alpha_src[n,hd] = <h[n,hd,:], att[hd,0:32]> ; alpha_dst uses att[hd,32:64]
// ---------------------------------------------------------------------------
__global__ void alpha_kernel(const float* __restrict__ h,
                             const float* __restrict__ att,
                             float* __restrict__ a_src,
                             float* __restrict__ a_dst) {
    int i = blockIdx.x * blockDim.x + threadIdx.x;
    if (i >= N_NODES * HEADS) return;
    int n  = i >> 2;
    int hd = i & 3;
    const float* hp = h + (size_t)n * HD + hd * OUT_DIM;
    const float* as = att + hd * (2 * OUT_DIM);
    const float* ad = as + OUT_DIM;
    float s1 = 0.f, s2 = 0.f;
#pragma unroll 8
    for (int d = 0; d < OUT_DIM; ++d) {
        float hv = hp[d];
        s1 += hv * as[d];
        s2 += hv * ad[d];
    }
    a_src[i] = s1;
    a_dst[i] = s2;
}

// ---------------------------------------------------------------------------
// Kernel 3: edge scatter. 32 lanes per edge; each lane owns 4 output dims
//   (float4 gather of h[src], 4 f32 atomics to out[dst]).
//   w = clip(ew,1e-8) * exp(leaky_relu(a_src[src]+a_dst[dst]))
//   All atomics land in L2 (working set << 192 MB).
// ---------------------------------------------------------------------------
__global__ __launch_bounds__(256) void edge_kernel(const long long* __restrict__ ei,
                                                   const float* __restrict__ ew,
                                                   const float* __restrict__ h,
                                                   const float* __restrict__ a_src,
                                                   const float* __restrict__ a_dst,
                                                   float* __restrict__ out,
                                                   float* __restrict__ att_sum) {
    long long t = (long long)blockIdx.x * blockDim.x + threadIdx.x;
    long long e = t >> 5;
    if (e >= N_EDGES) return;
    int lane = (int)(t & 31);          // covers dims [lane*4, lane*4+4)
    int head = lane >> 3;              // 8 lanes per head

    int src = (int)ei[e];
    int dst = (int)ei[N_EDGES + e];

    float al = a_src[src * HEADS + head] + a_dst[dst * HEADS + head];
    al = (al > 0.0f) ? al : 0.2f * al;                 // leaky_relu(0.2)
    float prior = ew[e];
    prior = (prior < 1e-8f) ? 1e-8f : prior;
    float w = prior * expf(al);

    const float4 hv = *(const float4*)(h + src * HD + lane * 4);   // b128 gather
    float* op = out + (size_t)dst * HD + lane * 4;
    atomicAdd(op + 0, w * hv.x);
    atomicAdd(op + 1, w * hv.y);
    atomicAdd(op + 2, w * hv.z);
    atomicAdd(op + 3, w * hv.w);
    if ((lane & 7) == 0) atomicAdd(&att_sum[dst * HEADS + head], w);
}

// ---------------------------------------------------------------------------
// Kernel 4: out /= (att_sum + 1e-8)
// ---------------------------------------------------------------------------
__global__ void finalize_kernel(float* __restrict__ out,
                                const float* __restrict__ att_sum) {
    int i = blockIdx.x * blockDim.x + threadIdx.x;
    if (i >= N_NODES * HD) return;
    int n    = i >> 7;
    int head = (i >> 5) & 3;
    out[i] = out[i] / (att_sum[n * HEADS + head] + 1e-8f);
}

// ---------------------------------------------------------------------------
extern "C" void kernel_launch(void* const* d_in, const int* in_sizes, int n_in,
                              void* d_out, int out_size, void* d_ws, size_t ws_size,
                              hipStream_t stream) {
    const float*     x   = (const float*)d_in[0];
    const long long* ei  = (const long long*)d_in[1];  // int64 in reference
    const float*     ew  = (const float*)d_in[2];
    const float*     W   = (const float*)d_in[3];
    const float*     att = (const float*)d_in[4];
    float*           out = (float*)d_out;

    // workspace layout
    char*      ws      = (char*)d_ws;
    float*     h       = (float*)ws;                                // N*128 f32
    float*     a_src   = (float*)(ws + (size_t)N_NODES * HD * 4);   // N*4
    float*     a_dst   = a_src + (size_t)N_NODES * HEADS;           // N*4
    float*     att_sum = a_dst + (size_t)N_NODES * HEADS;           // N*4
    _Float16*  Wh      = (_Float16*)(att_sum + (size_t)N_NODES * HEADS); // 128*128 f16

    // 0) zero accumulators
    {
        int blocks = (N_NODES * HD + 255) / 256;
        zero_kernel<<<blocks, 256, 0, stream>>>(out, att_sum);
    }
    // 0b) pack W to f16 (once)
    {
        int blocks = (HD * IN_DIM + 255) / 256;   // 64
        pack_w_kernel<<<blocks, 256, 0, stream>>>(W, Wh);
    }
    // 1) h = x @ W^T  (WMMA, K unrolled)
    {
        gemm_h_kernel<<<N_NODES / 16, 256, 0, stream>>>(x, Wh, h);  // 3125 blocks
    }
    // 2) per-node alphas
    {
        int blocks = (N_NODES * HEADS + 255) / 256;
        alpha_kernel<<<blocks, 256, 0, stream>>>(h, att, a_src, a_dst);
    }
    // 3) edge scatter
    {
        long long total = (long long)N_EDGES * 32;
        int blocks = (int)((total + 255) / 256);   // 100000
        edge_kernel<<<blocks, 256, 0, stream>>>(ei, ew, h, a_src, a_dst, out, att_sum);
    }
    // 4) normalize
    {
        int blocks = (N_NODES * HD + 255) / 256;
        finalize_kernel<<<blocks, 256, 0, stream>>>(out, att_sum);
    }
}
